// MultiheadAttention_81475529605226
// MI455X (gfx1250) — matrix operand order repaired and checked by reading, not compiled, not run on hardware
//
#include <hip/hip_runtime.h>
#include <hip/hip_bf16.h>
#include <cstdint>
#include <cstddef>

// ---------------------------------------------------------------------------
// MI455X (gfx1250) implementation: dual cross-attention fusion.
// All matmuls use v_wmma_f32_16x16x32_f16 (f16 in, f32 accumulate).
// K-tiles staged with global_load_async_to_lds_b128 (ASYNCcnt path).
// ---------------------------------------------------------------------------

typedef __attribute__((ext_vector_type(16))) _Float16 v16h;
typedef __attribute__((ext_vector_type(8)))  float    v8f;
typedef __attribute__((ext_vector_type(4)))  _Float16 h4;

#define EDIM 512
#define NHEAD 8
#define DH 64
#define NDLG 64
#define LMAX 256

static __device__ __forceinline__ v8f wmma16(v16h a, v16h b, v8f c) {
  // D = A(16x32 f16) * B(32x16 f16) + C(16x16 f32)
  return __builtin_amdgcn_wmma_f32_16x16x32_f16(false, a, false, b, (short)0, c,
                                                false, false);
}

// A-fragment k index for element i of v16h, given lane-half (contiguous k order, shared by A and B here):
// lanes 0-15 hold K {0..15}, lanes 16-31 hold K {16..31}
static __device__ __forceinline__ int a_kmap(int i, int half) {
  return i + 16 * half;
}

// CDNA5 async global->LDS copy, 16B per enabled lane (tracked by ASYNCcnt).
static __device__ __forceinline__ void async_load_b128(void* lds_ptr,
                                                       const void* gptr) {
  uint32_t ldsoff = (uint32_t)(uintptr_t)lds_ptr;  // LDS aperture: addr[31:0]
  unsigned long long ga = (unsigned long long)(uintptr_t)gptr;
  asm volatile("global_load_async_to_lds_b128 %0, %1, off"
               :: "v"(ldsoff), "v"(ga) : "memory");
}

// ---------------------------------------------------------------------------
// Prefix sum of dialog lengths -> token offsets
// ---------------------------------------------------------------------------
__global__ void offsets_kernel(const int* __restrict__ lengths,
                               int* __restrict__ off) {
  if (threadIdx.x == 0) {
    int acc = 0;
    for (int b = 0; b < NDLG; ++b) { off[b] = acc; acc += lengths[b]; }
    off[NDLG] = acc;
  }
}

// ---------------------------------------------------------------------------
// f32 -> f16 weight conversion
// ---------------------------------------------------------------------------
__global__ void cvt_f32_f16(const float* __restrict__ x,
                            _Float16* __restrict__ y, int n) {
  int i = blockIdx.x * blockDim.x + threadIdx.x;
  int stride = gridDim.x * blockDim.x;
  for (; i < n; i += stride) y[i] = (_Float16)x[i];
}

// ---------------------------------------------------------------------------
// LayerNorm both inputs, emit f16. One block (256 threads) per token row.
// ---------------------------------------------------------------------------
__global__ void __launch_bounds__(256)
ln_kernel(const float* __restrict__ x1, const float* __restrict__ x2,
          const float* __restrict__ g0, const float* __restrict__ b0,
          const float* __restrict__ g1, const float* __restrict__ b1,
          _Float16* __restrict__ y1, _Float16* __restrict__ y2, int N) {
  __shared__ float red[16];
  const int row = blockIdx.x;
  if (row >= N) return;
  const int tid = threadIdx.x;
  for (int which = 0; which < 2; ++which) {
    const float* x = which ? x2 : x1;
    const float* g = which ? g1 : g0;
    const float* bb = which ? b1 : b0;
    _Float16* y = which ? y2 : y1;
    float v0 = x[(size_t)row * EDIM + tid];
    float v1 = x[(size_t)row * EDIM + 256 + tid];
    float sum = v0 + v1;
    float sq = v0 * v0 + v1 * v1;
    for (int m = 1; m < 32; m <<= 1) {
      sum += __shfl_xor(sum, m, 32);
      sq  += __shfl_xor(sq,  m, 32);
    }
    if ((tid & 31) == 0) { red[tid >> 5] = sum; red[8 + (tid >> 5)] = sq; }
    __syncthreads();
    float ts = 0.f, tq = 0.f;
    for (int i = 0; i < 8; ++i) { ts += red[i]; tq += red[8 + i]; }
    float mean = ts * (1.0f / EDIM);
    float var  = tq * (1.0f / EDIM) - mean * mean;
    float rstd = rsqrtf(var + 1e-5f);
    y[(size_t)row * EDIM + tid] =
        (_Float16)((v0 - mean) * rstd * g[tid] + bb[tid]);
    y[(size_t)row * EDIM + 256 + tid] =
        (_Float16)((v1 - mean) * rstd * g[256 + tid] + bb[256 + tid]);
    __syncthreads();
  }
}

// ---------------------------------------------------------------------------
// WMMA GEMM: C[M x Nout] = A[M x 512](f16,row-major) * Wh[Nout x 512]^T + bias
// Block tile 128x64, K-step 64. 256 threads = 8 waves; wave w owns rows
// w*16..w*16+15 and four 16-wide N subtiles. Fragments are fully preloaded so
// the 8 WMMAs issue back-to-back. Next K-tile is prefetched (global_prefetch).
// Output: f16 (Ch, ldch) or f32 (Cf, ldcf, +coloff) for the final projection.
// ---------------------------------------------------------------------------
__global__ void __launch_bounds__(256)
gemm_wmma(const _Float16* __restrict__ A, const _Float16* __restrict__ Wh,
          const float* __restrict__ bias,
          _Float16* __restrict__ Ch, int ldch,
          float* __restrict__ Cf, int ldcf, int coloff, int M) {
  __shared__ _Float16 As[128 * 64];
  __shared__ _Float16 Bs[64 * 64];
  const int tid  = threadIdx.x;
  const int lane = tid & 31;
  const int wave = tid >> 5;
  const int half = lane >> 4;
  const int ln   = lane & 15;
  const int m0 = blockIdx.x * 128;
  const int n0 = blockIdx.y * 64;

  v8f zero = {};
  v8f acc[4];
  acc[0] = zero; acc[1] = zero; acc[2] = zero; acc[3] = zero;

  const int arow = tid >> 1, acol = (tid & 1) * 32;  // A staging: 4x float4
  const int brow = tid >> 2, bcol = (tid & 3) * 16;  // B staging: 2x float4

  for (int k0 = 0; k0 < EDIM; k0 += 64) {
    {  // stage A tile 128x64
      int gr = m0 + arow;
      float4 z = make_float4(0, 0, 0, 0);
      float4 d0 = z, d1 = z, d2 = z, d3 = z;
      if (gr < M) {
        const float4* p = (const float4*)(A + (size_t)gr * EDIM + k0 + acol);
        d0 = p[0]; d1 = p[1]; d2 = p[2]; d3 = p[3];
      }
      float4* q = (float4*)(As + arow * 64 + acol);
      q[0] = d0; q[1] = d1; q[2] = d2; q[3] = d3;
    }
    {  // stage B tile 64x64 (rows of Wh are output features; k contiguous)
      const float4* p =
          (const float4*)(Wh + (size_t)(n0 + brow) * EDIM + k0 + bcol);
      float4* q = (float4*)(Bs + brow * 64 + bcol);
      q[0] = p[0]; q[1] = p[1];
    }
    __syncthreads();

    // prefetch next K-tile into cache (gfx1250 global_prefetch_b8)
    if (k0 + 64 < EDIM) {
      if (m0 + arow < M)
        __builtin_prefetch(A + (size_t)(m0 + arow) * EDIM + k0 + 64 + acol, 0, 3);
      __builtin_prefetch(Wh + (size_t)(n0 + brow) * EDIM + k0 + 64 + bcol, 0, 3);
    }

    // preload fragments: 2 A ops (K 0..31 / 32..63), 8 B ops
    v16h a0, a1;
    const _Float16* ab = As + (wave * 16 + ln) * 64;
#pragma unroll
    for (int i = 0; i < 16; ++i) {
      int k = a_kmap(i, half);
      a0[i] = ab[k];
      a1[i] = ab[32 + k];
    }
    v16h bf[8];
#pragma unroll
    for (int j = 0; j < 4; ++j) {
      const _Float16* bb = Bs + (j * 16 + ln) * 64 + 16 * half;
#pragma unroll
      for (int i = 0; i < 16; ++i) {
        bf[2 * j][i]     = bb[i];
        bf[2 * j + 1][i] = bb[32 + i];
      }
    }
    // 8 back-to-back WMMAs
#pragma unroll
    for (int j = 0; j < 4; ++j) {
      acc[j] = wmma16(a0, bf[2 * j],     acc[j]);
      acc[j] = wmma16(a1, bf[2 * j + 1], acc[j]);
    }
    __syncthreads();
  }

#pragma unroll
  for (int j = 0; j < 4; ++j) {
    int col = n0 + j * 16 + ln;
    float bv = bias[col];
#pragma unroll
    for (int r = 0; r < 8; ++r) {
      int row = m0 + wave * 16 + r + 8 * half;
      if (row < M) {
        float v = acc[j][r] + bv;
        if (Ch) Ch[(size_t)row * ldch + col] = (_Float16)v;
        else    Cf[(size_t)row * ldcf + coloff + col] = v;
      }
    }
  }
}

// ---------------------------------------------------------------------------
// Flash-attention (one dialog, one head per block). 128 threads = 4 waves.
// LDS: K [LMAX][64] f16 (async-staged), V^T [64][LMAX] f16, per-wave P tile.
// Each wave processes 16-query tiles round-robin; online softmax in f32.
// ---------------------------------------------------------------------------
__global__ void __launch_bounds__(128)
attn_wmma(const _Float16* __restrict__ Qg, int ldq,
          const _Float16* __restrict__ Kg, int ldk,
          const _Float16* __restrict__ Vg, int ldv,
          _Float16* __restrict__ Og, const int* __restrict__ off) {
  extern __shared__ char smem[];
  _Float16* Ks = (_Float16*)smem;            // [LMAX][64]
  _Float16* Vt = Ks + LMAX * 64;             // [64][LMAX] (transposed V)
  _Float16* Ps = Vt + 64 * LMAX;             // 4 waves x [16][32]

  const int b = blockIdx.x >> 3;
  const int h = blockIdx.x & 7;
  const int o0  = off[b];
  const int len = off[b + 1] - o0;
  const int Lk  = (len + 31) & ~31;

  const int tid  = threadIdx.x;
  const int lane = tid & 31;
  const int wave = tid >> 5;
  const int half = lane >> 4;
  const int ln   = lane & 15;
  const _Float16 zh = (_Float16)0.f;

  // ---- zero-fill padded K rows (async loads have no OOB zero-fill) ----
  for (int i = tid; i < (Lk - len) * 16; i += 128) {
    int key = len + (i >> 4);
    int f   = (i & 15) * 4;
    h4 z = {zh, zh, zh, zh};
    *(h4*)(Ks + key * 64 + f) = z;
  }
  // ---- async global->LDS staging of valid K rows (8 x b128 per row) ----
  for (int i = tid; i < len * 8; i += 128) {
    int key = i >> 3;
    int seg = (i & 7) * 8;  // f16 elements; 16B segments
    async_load_b128(Ks + key * 64 + seg,
                    Kg + (size_t)(o0 + key) * ldk + h * DH + seg);
  }
  // ---- V transposed staging (vector global loads, scalar LDS stores) ----
  for (int idx = tid; idx < Lk * 16; idx += 128) {
    int key = idx >> 4;
    int f   = (idx & 15) * 4;
    h4 vd = {zh, zh, zh, zh};
    if (key < len)
      vd = *(const h4*)(Vg + (size_t)(o0 + key) * ldv + h * DH + f);
#pragma unroll
    for (int j = 0; j < 4; ++j) Vt[(f + j) * LMAX + key] = vd[j];
  }
  asm volatile("s_wait_asynccnt 0" ::: "memory");
  __syncthreads();

  _Float16* Pw = Ps + wave * (16 * 32);
  const float scale = 0.125f;  // 1/sqrt(64)
  const int nqt = (len + 15) >> 4;

  for (int t = wave; t < nqt; t += 4) {
    const int q0 = t * 16;
    // Q fragments (16x64 -> two 16x32 A operands), loaded once per tile
    v16h qa0, qa1;
    {
      bool ok = (q0 + ln) < len;
      const _Float16* qp = Qg + (size_t)(o0 + q0 + ln) * ldq + h * DH;
#pragma unroll
      for (int i = 0; i < 16; ++i) {
        int k = a_kmap(i, half);
        qa0[i] = ok ? qp[k]      : zh;
        qa1[i] = ok ? qp[32 + k] : zh;
      }
    }
    float mrow[8], srow[8];
#pragma unroll
    for (int r = 0; r < 8; ++r) { mrow[r] = -1e30f; srow[r] = 0.f; }
    v8f zero = {};
    v8f o[4];
    o[0] = zero; o[1] = zero; o[2] = zero; o[3] = zero;

    for (int kc = 0; kc < Lk; kc += 32) {
      // scores for two 16-key subtiles (f32 accumulate over Dh=64)
      v16h b0, b1, b2, b3;
      const _Float16* kb0 = Ks + (kc + ln) * 64 + 16 * half;
      const _Float16* kb1 = Ks + (kc + 16 + ln) * 64 + 16 * half;
#pragma unroll
      for (int i = 0; i < 16; ++i) {
        b0[i] = kb0[i];      b1[i] = kb0[32 + i];
        b2[i] = kb1[i];      b3[i] = kb1[32 + i];
      }
      v8f s0 = zero, s1 = zero;
      s0 = wmma16(qa0, b0, s0);
      s0 = wmma16(qa1, b1, s0);
      s1 = wmma16(qa0, b2, s1);
      s1 = wmma16(qa1, b3, s1);

      const bool ok0 = (kc + ln) < len;
      const bool ok1 = (kc + 16 + ln) < len;
#pragma unroll
      for (int r = 0; r < 8; ++r) {
        s0[r] = ok0 ? s0[r] * scale : -1e30f;
        s1[r] = ok1 ? s1[r] * scale : -1e30f;
      }
      // online softmax: row reductions span 16 lanes (wave32, width=16)
#pragma unroll
      for (int r = 0; r < 8; ++r) {
        float cm = fmaxf(s0[r], s1[r]);
        for (int m2 = 1; m2 < 16; m2 <<= 1)
          cm = fmaxf(cm, __shfl_xor(cm, m2, 16));
        float mn = fmaxf(mrow[r], cm);
        float alpha = __expf(mrow[r] - mn);
        float p0 = __expf(s0[r] - mn);
        float p1 = __expf(s1[r] - mn);
        float ps = p0 + p1;
        for (int m2 = 1; m2 < 16; m2 <<= 1) ps += __shfl_xor(ps, m2, 16);
        srow[r] = srow[r] * alpha + ps;
        mrow[r] = mn;
        s0[r] = p0; s1[r] = p1;
#pragma unroll
        for (int j = 0; j < 4; ++j) o[j][r] *= alpha;
      }
      // C-layout -> A-layout via per-wave LDS (DS is in-order per wave;
      // s_wait_dscnt flushes the split DS counter before readback)
#pragma unroll
      for (int r = 0; r < 8; ++r) {
        Pw[(r + 8 * half) * 32 + ln]      = (_Float16)s0[r];
        Pw[(r + 8 * half) * 32 + 16 + ln] = (_Float16)s1[r];
      }
      asm volatile("s_wait_dscnt 0" ::: "memory");
      v16h pa;
#pragma unroll
      for (int i = 0; i < 16; ++i) pa[i] = Pw[ln * 32 + a_kmap(i, half)];
      // O += P(16x32) @ V(32x64) via transposed-V B operands
#pragma unroll
      for (int j = 0; j < 4; ++j) {
        v16h vb;
        const _Float16* vbase = Vt + (j * 16 + ln) * LMAX + kc + 16 * half;
#pragma unroll
        for (int i = 0; i < 16; ++i) vb[i] = vbase[i];
        o[j] = wmma16(pa, vb, o[j]);
      }
    }
    // normalize + store (flat layout, only valid rows)
#pragma unroll
    for (int j = 0; j < 4; ++j) {
#pragma unroll
      for (int r = 0; r < 8; ++r) {
        int row = q0 + r + 8 * half;
        if (row < len) {
          Og[(size_t)(o0 + row) * EDIM + h * DH + j * 16 + ln] =
              (_Float16)(o[j][r] / srow[r]);
        }
      }
    }
  }
}

// ---------------------------------------------------------------------------
// Host-side orchestration
// ---------------------------------------------------------------------------
extern "C" void kernel_launch(void* const* d_in, const int* in_sizes, int n_in,
                              void* d_out, int out_size, void* d_ws,
                              size_t ws_size, hipStream_t stream) {
  const float* x1       = (const float*)d_in[0];
  const float* x2       = (const float*)d_in[1];
  const int*   lengths  = (const int*)d_in[2];
  const float* ln0_g    = (const float*)d_in[3];
  const float* ln0_b    = (const float*)d_in[4];
  const float* ln1_g    = (const float*)d_in[5];
  const float* ln1_b    = (const float*)d_in[6];
  const float* ld_in_w  = (const float*)d_in[7];
  const float* ld_in_b  = (const float*)d_in[8];
  const float* ld_out_w = (const float*)d_in[9];
  const float* ld_out_b = (const float*)d_in[10];
  const float* dl_in_w  = (const float*)d_in[11];
  const float* dl_in_b  = (const float*)d_in[12];
  const float* dl_out_w = (const float*)d_in[13];
  const float* dl_out_b = (const float*)d_in[14];

  const int N = in_sizes[0] / EDIM;

  // ---- carve workspace (f16 activations + f16 weights + offsets) ----
  char* p = (char*)d_ws;
  auto alloc = [&](size_t bytes) -> char* {
    char* r = p;
    p += (bytes + 255) & ~(size_t)255;
    return r;
  };
  _Float16* x1n = (_Float16*)alloc((size_t)N * EDIM * 2);
  _Float16* x2n = (_Float16*)alloc((size_t)N * EDIM * 2);
  _Float16* Q1  = (_Float16*)alloc((size_t)N * 512 * 2);
  _Float16* KV1 = (_Float16*)alloc((size_t)N * 1024 * 2);
  _Float16* Q2  = (_Float16*)alloc((size_t)N * 512 * 2);
  _Float16* KV2 = (_Float16*)alloc((size_t)N * 1024 * 2);
  _Float16* O1  = (_Float16*)alloc((size_t)N * 512 * 2);
  _Float16* O2  = (_Float16*)alloc((size_t)N * 512 * 2);
  _Float16* Wldin  = (_Float16*)alloc((size_t)1536 * 512 * 2);
  _Float16* Wdlin  = (_Float16*)alloc((size_t)1536 * 512 * 2);
  _Float16* Wldout = (_Float16*)alloc((size_t)512 * 512 * 2);
  _Float16* Wdlout = (_Float16*)alloc((size_t)512 * 512 * 2);
  int* off = (int*)alloc((NDLG + 1) * sizeof(int));

  offsets_kernel<<<1, 32, 0, stream>>>(lengths, off);
  ln_kernel<<<N, 256, 0, stream>>>(x1, x2, ln0_g, ln0_b, ln1_g, ln1_b,
                                   x1n, x2n, N);
  cvt_f32_f16<<<1024, 256, 0, stream>>>(ld_in_w,  Wldin,  1536 * 512);
  cvt_f32_f16<<<1024, 256, 0, stream>>>(dl_in_w,  Wdlin,  1536 * 512);
  cvt_f32_f16<<<512,  256, 0, stream>>>(ld_out_w, Wldout, 512 * 512);
  cvt_f32_f16<<<512,  256, 0, stream>>>(dl_out_w, Wdlout, 512 * 512);

  dim3 blk(256);
  dim3 g512((N + 127) / 128, 8);    // Nout = 512
  dim3 g1024((N + 127) / 128, 16);  // Nout = 1024

  // Input projections (torch MHA: q from query, k/v from key/value input)
  gemm_wmma<<<g512,  blk, 0, stream>>>(x1n, Wldin,              ld_in_b,
                                       Q1, 512, nullptr, 0, 0, N);
  gemm_wmma<<<g1024, blk, 0, stream>>>(x2n, Wldin + 512 * 512,  ld_in_b + 512,
                                       KV1, 1024, nullptr, 0, 0, N);
  gemm_wmma<<<g512,  blk, 0, stream>>>(x2n, Wdlin,              dl_in_b,
                                       Q2, 512, nullptr, 0, 0, N);
  gemm_wmma<<<g1024, blk, 0, stream>>>(x1n, Wdlin + 512 * 512,  dl_in_b + 512,
                                       KV2, 1024, nullptr, 0, 0, N);

  const size_t attn_lds = (size_t)(LMAX * 64 + 64 * LMAX + 4 * 16 * 32) * 2;
  attn_wmma<<<NDLG * NHEAD, 128, attn_lds, stream>>>(Q1, 512, KV1, 1024,
                                                     KV1 + 512, 1024, O1, off);
  attn_wmma<<<NDLG * NHEAD, 128, attn_lds, stream>>>(Q2, 512, KV2, 1024,
                                                     KV2 + 512, 1024, O2, off);

  // Output projections write f32 directly into concatenated [N,1024] output
  gemm_wmma<<<g512, blk, 0, stream>>>(O1, Wldout, ld_out_b,
                                      nullptr, 0, (float*)d_out, 1024, 0, N);
  gemm_wmma<<<g512, blk, 0, stream>>>(O2, Wdlout, dl_out_b,
                                      nullptr, 0, (float*)d_out, 1024, 512, N);
}
